// RNN_81406810128871
// MI455X (gfx1250) — matrix operand order repaired and checked by reading, not compiled, not run on hardware
//
#include <hip/hip_runtime.h>
#include <hip/hip_bf16.h>
#include <math.h>

// Sizes from reference: T=2048, B=64, I=512, H=512 (fp32 in/out).
#define T_STEPS 2048
#define BATCH   64
#define HID     512
#define HSTATE  (BATCH * HID)        // 32768 elements of h
#define NWG_REC 16
#define COLS_PER_WG (HID / NWG_REC)  // 32 output columns per recurrent WG

typedef _Float16 f16_t;
typedef __attribute__((ext_vector_type(16))) _Float16 v16h;
typedef __attribute__((ext_vector_type(8)))  float    v8f;

union ABFrag { v16h v; uint4 q[2]; };

__device__ __forceinline__ v8f wmma_f16(const ABFrag& a, const ABFrag& b, v8f c) {
  // (neg_a, A, neg_b, B, c_mod, C, reuse_a, reuse_b)
  return __builtin_amdgcn_wmma_f32_16x16x32_f16(false, a.v, false, b.v, (short)0, c,
                                                false, false);
}

__device__ __forceinline__ void split_f32(float v, f16_t& hi, f16_t& lo) {
  hi = (f16_t)v;
  lo = (f16_t)(v - (float)hi);
}

// ---------------------------------------------------------------------------
// Kernel 0: split weights into f16 hi/lo planes, fold biases, zero h0, reset
// the grid-barrier counter (must happen every launch for graph replay).
// ---------------------------------------------------------------------------
__global__ __launch_bounds__(256) void init_kernel(
    const float* __restrict__ Wih, const float* __restrict__ Whh,
    const float* __restrict__ bih, const float* __restrict__ bhh,
    f16_t* __restrict__ Wih_hi, f16_t* __restrict__ Wih_lo,
    f16_t* __restrict__ Whh_hi, f16_t* __restrict__ Whh_lo,
    float* __restrict__ bsum, f16_t* __restrict__ hbuf0,
    unsigned* __restrict__ ctr) {
  const int tid = blockIdx.x * blockDim.x + threadIdx.x;
  const int n   = gridDim.x * blockDim.x;
  for (int i = tid; i < HID * HID; i += n) {
    split_f32(Wih[i], Wih_hi[i], Wih_lo[i]);
    split_f32(Whh[i], Whh_hi[i], Whh_lo[i]);
  }
  for (int i = tid; i < 2 * HSTATE; i += n) hbuf0[i] = (f16_t)0.0f;  // h0 hi+lo
  for (int i = tid; i < HID; i += n) bsum[i] = bih[i] + bhh[i];
  if (tid == 0) *ctr = 0u;
}

// ---------------------------------------------------------------------------
// Kernel 1: xproj[m, n] = sum_k x[m,k] * W_ih[n,k] + bsum[n]   (into d_out)
// One WG per 32 M-rows: stage x stripe once as f16 hi/lo in LDS (64 KB),
// sweep all N=512.  Each K-chunk does 3 WMMAs (hi*hi + hi*lo + lo*hi) for
// ~fp32-faithful accuracy out of the f16 matrix pipe.
// ---------------------------------------------------------------------------
__global__ __launch_bounds__(256) void xproj_kernel(
    const float* __restrict__ x,
    const f16_t* __restrict__ Wih_hi, const f16_t* __restrict__ Wih_lo,
    const float* __restrict__ bsum, float* __restrict__ out) {
  __shared__ f16_t Ax_hi[32 * HID];  // 32 KB
  __shared__ f16_t Ax_lo[32 * HID];  // 32 KB
  const int tid = threadIdx.x;
  const size_t rowBase = (size_t)blockIdx.x * 32;

  // Stage 32x512 f32 -> f16 hi/lo planes (x read exactly once from HBM).
  const float4* xs = (const float4*)(x + rowBase * HID);
  for (int i = tid; i < 32 * (HID / 4); i += 256) {
    float4 f = xs[i];
    int o = i * 4;
    split_f32(f.x, Ax_hi[o + 0], Ax_lo[o + 0]);
    split_f32(f.y, Ax_hi[o + 1], Ax_lo[o + 1]);
    split_f32(f.z, Ax_hi[o + 2], Ax_lo[o + 2]);
    split_f32(f.w, Ax_hi[o + 3], Ax_lo[o + 3]);
  }
  __syncthreads();

  const int wid = tid >> 5, lane = tid & 31;
  const int mt      = wid & 1;          // 2 M-tiles of 16 rows
  const int ntStart = (wid >> 1) * 8;   // 4 groups x 8 N-tiles = 32 tiles
  const int lhalf   = lane >> 4;
  const int l15     = lane & 15;
  const int rowA    = mt * 16 + l15;

  for (int nt = ntStart; nt < ntStart + 8; ++nt) {
    const int col = nt * 16 + l15;      // output column this lane owns in C
    const float bs = bsum[col];
    v8f c;
#pragma unroll
    for (int r = 0; r < 8; ++r) c[r] = bs;

    const f16_t* Bh = Wih_hi + (size_t)col * HID;  // B column n = W_ih row n
    const f16_t* Bl = Wih_lo + (size_t)col * HID;
#pragma unroll 4
    for (int kc = 0; kc < HID / 32; ++kc) {
      ABFrag ah, al, bh, bl;
      const int kbA = kc * 32 + lhalf * 16;  // A: 16 contiguous K
      const int oA  = rowA * HID + kbA;
      ah.q[0] = *(const uint4*)(Ax_hi + oA);
      ah.q[1] = *(const uint4*)(Ax_hi + oA + 8);
      al.q[0] = *(const uint4*)(Ax_lo + oA);
      al.q[1] = *(const uint4*)(Ax_lo + oA + 8);
      const int kbB = kc * 32 + lhalf * 16;  // B: 16 contiguous K
      bh.q[0] = *(const uint4*)(Bh + kbB);
      bh.q[1] = *(const uint4*)(Bh + kbB + 8);
      bl.q[0] = *(const uint4*)(Bl + kbB);
      bl.q[1] = *(const uint4*)(Bl + kbB + 8);
      c = wmma_f16(ah, bh, c);
      c = wmma_f16(ah, bl, c);
      c = wmma_f16(al, bh, c);
    }
#pragma unroll
    for (int r = 0; r < 8; ++r) {
      const int m = mt * 16 + r + 8 * lhalf;
      out[(rowBase + m) * HID + col] = c[r];
    }
  }
}

// ---------------------------------------------------------------------------
// Kernel 2: persistent recurrent scan. 16 WGs, each owns 32 output columns and
// keeps its W_hh slice as f16 hi/lo (64 KB LDS) for all 2048 steps.  The h
// state lives in L2 as double-buffered f16 hi/lo planes (state is effectively
// fp32), with one grid barrier per step.  out[t] holds xproj(t) on entry and
// is overwritten in place with h_t.  Next step's xproj tile is prefetched
// before the barrier to hide its L2 latency from the serial chain.
// ---------------------------------------------------------------------------
__global__ __launch_bounds__(256) void rnn_scan_kernel(
    const f16_t* __restrict__ Whh_hi, const f16_t* __restrict__ Whh_lo,
    float* __restrict__ out, f16_t* __restrict__ hbuf,
    unsigned* __restrict__ ctr) {
  __shared__ f16_t Ws_hi[COLS_PER_WG * HID];  // 32 KB
  __shared__ f16_t Ws_lo[COLS_PER_WG * HID];  // 32 KB
  const int tid = threadIdx.x;
  const int g   = blockIdx.x;

  {  // load this WG's W_hh slice once (rows g*32 .. g*32+31), hi+lo
    const uint4* sh = (const uint4*)(Whh_hi + (size_t)g * COLS_PER_WG * HID);
    const uint4* sl = (const uint4*)(Whh_lo + (size_t)g * COLS_PER_WG * HID);
    uint4* dh = (uint4*)Ws_hi;
    uint4* dl = (uint4*)Ws_lo;
    for (int i = tid; i < COLS_PER_WG * HID / 8; i += 256) {
      dh[i] = sh[i];
      dl[i] = sl[i];
    }
  }
  __syncthreads();

  const int wid = tid >> 5, lane = tid & 31;
  const int mt    = wid >> 1;               // 4 M-tiles (B=64)
  const int nt    = wid & 1;                // 2 N-tiles (32 cols)
  const int lhalf = lane >> 4;
  const int l15   = lane & 15;
  const int rowA  = mt * 16 + l15;
  const int colL  = nt * 16 + l15;          // 0..31 within slice
  const int colG  = g * COLS_PER_WG + colL; // 0..511 global

  // Preload xproj tile for t=0.
  v8f xp;
#pragma unroll
  for (int r = 0; r < 8; ++r)
    xp[r] = out[(mt * 16 + r + 8 * lhalf) * HID + colG];

  for (int t = 0; t < T_STEPS; ++t) {
    // Double-buffered h planes: buffer (t&1) is read, ((t+1)&1) is written.
    const f16_t* hr_hi = hbuf + (size_t)(t & 1) * (2 * HSTATE);
    const f16_t* hr_lo = hr_hi + HSTATE;
    f16_t* hw_hi = hbuf + (size_t)((t + 1) & 1) * (2 * HSTATE);
    f16_t* hw_lo = hw_hi + HSTATE;
    float* outT  = out + (size_t)t * HSTATE;

    v8f c = xp;  // xproj already includes both biases
#pragma unroll 4
    for (int kc = 0; kc < HID / 32; ++kc) {
      ABFrag ah, al, bh, bl;
      const int kbA = kc * 32 + lhalf * 16;
      const int oA  = rowA * HID + kbA;
      ah.q[0] = *(const uint4*)(hr_hi + oA);
      ah.q[1] = *(const uint4*)(hr_hi + oA + 8);
      al.q[0] = *(const uint4*)(hr_lo + oA);
      al.q[1] = *(const uint4*)(hr_lo + oA + 8);
      const int kbB = kc * 32 + lhalf * 16;
      const int oB  = colL * HID + kbB;
      bh.q[0] = *(const uint4*)(Ws_hi + oB);
      bh.q[1] = *(const uint4*)(Ws_hi + oB + 8);
      bl.q[0] = *(const uint4*)(Ws_lo + oB);
      bl.q[1] = *(const uint4*)(Ws_lo + oB + 8);
      c = wmma_f16(ah, bh, c);
      c = wmma_f16(ah, bl, c);
      c = wmma_f16(al, bh, c);
    }

#pragma unroll
    for (int r = 0; r < 8; ++r) {
      const int m  = mt * 16 + r + 8 * lhalf;
      const float h = tanhf(c[r]);
      outT[m * HID + colG] = h;  // final output for step t (f32)
      f16_t hh, hl;
      split_f32(h, hh, hl);      // next-step A operand, fp32-faithful
      hw_hi[m * HID + colG] = hh;
      hw_lo[m * HID + colG] = hl;
    }

    // Prefetch next step's xproj tile before the barrier (hides L2 latency).
    if (t + 1 < T_STEPS) {
      const float* outN = out + (size_t)(t + 1) * HSTATE;
#pragma unroll
      for (int r = 0; r < 8; ++r)
        xp[r] = outN[(mt * 16 + r + 8 * lhalf) * HID + colG];
    }

    // Grid-wide barrier: release my slice, wait for all 16 WGs.
    __threadfence();
    __syncthreads();
    if (tid == 0) {
      atomicAdd(ctr, 1u);
      const unsigned target = (unsigned)(NWG_REC * (t + 1));
      while (atomicAdd(ctr, 0u) < target) { __builtin_amdgcn_s_sleep(2); }
    }
    __syncthreads();
    __threadfence();
  }
}

// ---------------------------------------------------------------------------
// Host-side launch. Inputs: x, W_ih, W_hh, b_ih, b_hh (all f32).
// ws layout (bytes):
//   [Wih_hi 512K][Wih_lo 512K][Whh_hi 512K][Whh_lo 512K]
//   [bsum 2K][hbuf 2 bufs x (hi+lo) x 64K = 256K][ctr 4B]        ~2.4 MB
// ---------------------------------------------------------------------------
extern "C" void kernel_launch(void* const* d_in, const int* in_sizes, int n_in,
                              void* d_out, int out_size, void* d_ws, size_t ws_size,
                              hipStream_t stream) {
  const float* x   = (const float*)d_in[0];
  const float* Wih = (const float*)d_in[1];
  const float* Whh = (const float*)d_in[2];
  const float* bih = (const float*)d_in[3];
  const float* bhh = (const float*)d_in[4];
  float* out = (float*)d_out;

  char* ws = (char*)d_ws;
  const size_t WB = (size_t)HID * HID * sizeof(f16_t);  // 512 KB per plane
  f16_t*    Wih_hi = (f16_t*)(ws);
  f16_t*    Wih_lo = (f16_t*)(ws + WB);
  f16_t*    Whh_hi = (f16_t*)(ws + 2 * WB);
  f16_t*    Whh_lo = (f16_t*)(ws + 3 * WB);
  float*    bsum   = (float*)(ws + 4 * WB);
  f16_t*    hbuf   = (f16_t*)(ws + 4 * WB + 2048);
  unsigned* ctr    = (unsigned*)(ws + 4 * WB + 2048 + 4 * HSTATE * sizeof(f16_t));

  init_kernel<<<256, 256, 0, stream>>>(Wih, Whh, bih, bhh,
                                       Wih_hi, Wih_lo, Whh_hi, Whh_lo,
                                       bsum, hbuf, ctr);
  xproj_kernel<<<(T_STEPS * BATCH) / 32, 256, 0, stream>>>(x, Wih_hi, Wih_lo,
                                                           bsum, out);
  rnn_scan_kernel<<<NWG_REC, 256, 0, stream>>>(Whh_hi, Whh_lo, out, hbuf, ctr);
}
